// FastTransformerDecoder_51513837748452
// MI455X (gfx1250) — matrix operand order repaired and checked by reading, not compiled, not run on hardware
//
#include <hip/hip_runtime.h>

// ---------------------------------------------------------------------------
// CDNA5 (gfx1250) FastTransformer decoder: Performer/FAVOR causal linear attn.
// GEMMs in f16 WMMA (f32 accumulate), attention scan + feature map in fp32.
// Async global->LDS staging (ASYNCcnt) in attention; global_prefetch in GEMM.
// ---------------------------------------------------------------------------

typedef _Float16 v16h __attribute__((ext_vector_type(16)));
typedef _Float16 v8h  __attribute__((ext_vector_type(8)));
typedef _Float16 v4h  __attribute__((ext_vector_type(4)));
typedef float    v8f  __attribute__((ext_vector_type(8)));

#define LN_EPS   1e-5f
#define ATTN_EPS 1e-6f

__device__ __forceinline__ void wait_asynccnt0() {
#if __has_builtin(__builtin_amdgcn_s_wait_asynccnt)
    __builtin_amdgcn_s_wait_asynccnt(0);
#else
    asm volatile("s_wait_asynccnt 0x0" ::: "memory");
#endif
}

// Per-lane async copy: 16 bytes global -> LDS, tracked by ASYNCcnt.
__device__ __forceinline__ void async_g2l_b128(unsigned lds_off, const void* gptr) {
    asm volatile("global_load_async_to_lds_b128 %0, %1, off"
                 :: "v"(lds_off), "v"((unsigned long long)(uintptr_t)gptr)
                 : "memory");
}

// ---------------------------------------------------------------------------
// Tiled GEMM: C[M,N] = A[M,K] @ W[K,N] + bias (+ optional ReLU)
// Block: 256 threads = 8 waves (2x4).  Tile: 128x128, K-step 32.
// Per wave: 64x32 output = 4x2 accumulators of v_wmma_f32_16x16x32_f16.
// ---------------------------------------------------------------------------
__global__ __launch_bounds__(256)
void gemm_wmma(const float* __restrict__ A, const float* __restrict__ W,
               const float* __restrict__ bias, float* __restrict__ C,
               int M, int N, int K, int relu)
{
    constexpr int BM = 128, BN = 128, BK = 32;
    __shared__ alignas(16) _Float16 As[2][BM][BK + 8];   // row-major [m][k]
    __shared__ alignas(16) _Float16 Bs[2][BN][BK + 8];   // transposed [n][k]

    const int tid  = threadIdx.x;
    const int bm   = blockIdx.y, bn = blockIdx.x;
    const int wave = tid >> 5,   lane = tid & 31;
    const int wm   = wave >> 2,  wn   = wave & 3;        // 2 x 4 wave grid
    const int g    = lane >> 4,  lr   = lane & 15;       // half-wave group / row

    float4 aReg[4], bReg[4];

    auto loadTiles = [&](int kt) {
#pragma unroll
        for (int i = 0; i < 4; ++i) {                    // A: 128x32 floats
            int e = tid + i * 256;                       // 1024 float4s
            int r = e >> 3, c = (e & 7) << 2;
            aReg[i] = *(const float4*)&A[(size_t)(bm * BM + r) * K + kt * BK + c];
        }
#pragma unroll
        for (int i = 0; i < 4; ++i) {                    // W: 32x128 floats
            int e = tid + i * 256;
            int r = e >> 5, c = (e & 31) << 2;
            bReg[i] = *(const float4*)&W[(size_t)(kt * BK + r) * N + bn * BN + c];
        }
    };
    auto storeTiles = [&](int buf) {
#pragma unroll
        for (int i = 0; i < 4; ++i) {
            int e = tid + i * 256;
            int r = e >> 3, c = (e & 7) << 2;
            v4h h4 = { (_Float16)aReg[i].x, (_Float16)aReg[i].y,
                       (_Float16)aReg[i].z, (_Float16)aReg[i].w };
            *(v4h*)&As[buf][r][c] = h4;
        }
#pragma unroll
        for (int i = 0; i < 4; ++i) {                    // transpose into Bs[n][k]
            int e = tid + i * 256;
            int r = e >> 5, c = (e & 31) << 2;
            Bs[buf][c + 0][r] = (_Float16)bReg[i].x;
            Bs[buf][c + 1][r] = (_Float16)bReg[i].y;
            Bs[buf][c + 2][r] = (_Float16)bReg[i].z;
            Bs[buf][c + 3][r] = (_Float16)bReg[i].w;
        }
    };

    const v8f zero8 = {0.f, 0.f, 0.f, 0.f, 0.f, 0.f, 0.f, 0.f};
    v8f acc[4][2];
#pragma unroll
    for (int i = 0; i < 4; ++i)
#pragma unroll
        for (int j = 0; j < 2; ++j) acc[i][j] = zero8;

    const int nk = K / BK;
    loadTiles(0);
    storeTiles(0);
    __syncthreads();

    for (int kt = 0; kt < nk; ++kt) {
        const int buf = kt & 1;
        if (kt + 1 < nk) loadTiles(kt + 1);
        if (kt + 2 < nk) {                    // global_prefetch_b8 of tile kt+2
            __builtin_prefetch(&A[(size_t)(bm * BM + (tid >> 3)) * K
                                  + (kt + 2) * BK + ((tid & 7) << 2)], 0, 3);
            __builtin_prefetch(&W[(size_t)((kt + 2) * BK + (tid >> 5)) * N
                                  + bn * BN + ((tid & 31) << 2)], 0, 3);
        }

        // A fragment (ISA 7.12.2, 16-bit A 16x32): lane = g*16 + m,
        // halves h<8 -> k = 8g+h ; h>=8 -> k = 16+8g+(h-8).  Two b128 reads.
        v16h af[4];
#pragma unroll
        for (int i = 0; i < 4; ++i) {
            int row = wm * 64 + i * 16 + lr;
            v8h lo = *(const v8h*)&As[buf][row][8 * g];
            v8h hi = *(const v8h*)&As[buf][row][16 + 8 * g];
            af[i] = __builtin_shufflevector(lo, hi, 0, 1, 2, 3, 4, 5, 6, 7,
                                            8, 9, 10, 11, 12, 13, 14, 15);
        }
        // B fragment (dense 32x16): lane = g*16 + n, halves h<8 -> k = 8g+h ; h>=8 -> k = 16+8g+(h-8).
        v16h bfm[2];
#pragma unroll
        for (int j = 0; j < 2; ++j) {
            int col = wn * 32 + j * 16 + lr;
            v8h lo = *(const v8h*)&Bs[buf][col][8 * g];
            v8h hi = *(const v8h*)&Bs[buf][col][16 + 8 * g];
            bfm[j] = __builtin_shufflevector(lo, hi, 0, 1, 2, 3, 4, 5, 6, 7,
                                             8, 9, 10, 11, 12, 13, 14, 15);
        }
#pragma unroll
        for (int i = 0; i < 4; ++i)
#pragma unroll
            for (int j = 0; j < 2; ++j)
                acc[i][j] = __builtin_amdgcn_wmma_f32_16x16x32_f16(
                    false, af[i], false, bfm[j], (short)0, acc[i][j], false, false);

        if (kt + 1 < nk) storeTiles(buf ^ 1);
        __syncthreads();
    }

    // Epilogue: C/D layout — VGPR e holds M = e + 8g, N = lane&15.
    const int mbase = bm * BM + wm * 64;
    const int nbase = bn * BN + wn * 32;
#pragma unroll
    for (int i = 0; i < 4; ++i) {
#pragma unroll
        for (int j = 0; j < 2; ++j) {
            int col = nbase + j * 16 + lr;
            float bvv = bias[col];
#pragma unroll
            for (int e = 0; e < 8; ++e) {
                int row = mbase + i * 16 + e + 8 * g;
                float v = acc[i][j][e] + bvv;
                if (relu) v = v > 0.f ? v : 0.f;
                C[(size_t)row * N + col] = v;
            }
        }
    }
}

// ---------------------------------------------------------------------------
// FAVOR feature map, in place.  One 64-thread block per (b,s,h) row.
// phi = [exp(u - h), exp(-u - h)] * m^-1/2 ; u = xs@omega, h = ||xs||^2/2,
// xs = x * dh^-1/4.  dh = 64, m = 64.
// ---------------------------------------------------------------------------
__global__ __launch_bounds__(64)
void favor_kernel(float* __restrict__ qk, const float* __restrict__ omega)
{
    __shared__ float xs[64];
    const int t = threadIdx.x;
    float* base = qk + (size_t)blockIdx.x * 64;
    const float scale = 0.35355339059327373f;   // 64^-0.25
    xs[t] = base[t] * scale;
    __syncthreads();

    float hh = 0.f;
#pragma unroll
    for (int d = 0; d < 64; ++d) hh += xs[d] * xs[d];
    hh *= 0.5f;

    if (t < 32) {
        float u = 0.f;
        for (int d = 0; d < 64; ++d) u += xs[d] * omega[d * 32 + t];
        float po = expf(u - hh) * 0.125f;       // m^-0.5 = 1/8
        float pn = expf(-u - hh) * 0.125f;
        base[t]      = po;
        base[32 + t] = pn;
    }
}

// ---------------------------------------------------------------------------
// Causal linear attention, chunked recurrence.  One block per (b,h).
// State S[64x64], z[64] in LDS fp32; chunk = 64 timesteps.
// Chunk staging uses GLOBAL_LOAD_ASYNC_TO_LDS_B128 (ASYNCcnt).
// ---------------------------------------------------------------------------
__global__ __launch_bounds__(256)
void causal_linear_attn(const float* __restrict__ Qf, const float* __restrict__ Kf,
                        const float* __restrict__ V, float* __restrict__ Out)
{
    constexpr int S = 1024, H = 16, CHK = 64;
    constexpr int P = 68;                      // 16B-aligned row stride (async dst)
    __shared__ float Sst[64][65];
    __shared__ alignas(16) float Qc[64][P], Kc[64][P], Vc[64][P];
    __shared__ float Am[64][65];
    __shared__ float zs[64], den[64], dinter[64];

    const int bh = blockIdx.x;
    const int b = bh >> 4, h = bh & 15;
    const int tid = threadIdx.x;
    const int d  = tid & 63;          // owned column
    const int gq = tid >> 6;          // row group 0..3 (16 rows each)

    for (int e = tid; e < 64 * 64; e += 256) Sst[e >> 6][e & 63] = 0.f;
    if (tid < 64) zs[tid] = 0.f;
    __syncthreads();

    const size_t headBase = ((size_t)b * S * H + h) * 64;   // + t*H*64 + m

    for (int cc = 0; cc < S / CHK; ++cc) {
        const int t0 = cc * CHK;

        // Async global->LDS chunk staging: 64x64 floats per matrix, 16B/lane.
#pragma unroll
        for (int s4 = 0; s4 < 4; ++s4) {
            int e4 = tid + s4 * 256;                   // float4 index 0..1023
            int i  = e4 >> 4;                          // chunk row 0..63
            int m4 = (e4 & 15) << 2;                   // col 0,4,..,60
            size_t gi = headBase + (size_t)(t0 + i) * (H * 64) + m4;
            async_g2l_b128((unsigned)(uintptr_t)&Qc[i][m4], Qf + gi);
            async_g2l_b128((unsigned)(uintptr_t)&Kc[i][m4], Kf + gi);
            async_g2l_b128((unsigned)(uintptr_t)&Vc[i][m4], V  + gi);
        }
        wait_asynccnt0();
        __syncthreads();

        // Phase A: inter-chunk numerator partials + full A = Qc Kc^T column d.
        float accI[16];
#pragma unroll
        for (int r = 0; r < 16; ++r) {
            int i = gq * 16 + r;
            float s = 0.f, a = 0.f;
            for (int m = 0; m < 64; ++m) {
                s += Qc[i][m] * Sst[m][d];
                a += Qc[i][m] * Kc[d][m];
            }
            accI[r] = s;
            Am[i][d] = a;
        }
        if (tid < 64) {
            float s = 0.f;
            for (int m = 0; m < 64; ++m) s += Qc[tid][m] * zs[m];
            dinter[tid] = s;
        }
        __syncthreads();

        // Denominator: q.z_prev + causal prefix of A row + eps.
        if (tid < 64) {
            float s = dinter[tid];
            for (int j = 0; j <= tid; ++j) s += Am[tid][j];
            den[tid] = s + ATTN_EPS;
        }
        __syncthreads();

        // Phase B: intra-chunk causal A@V, normalize, write.
        for (int r = 0; r < 16; ++r) {
            int i = gq * 16 + r;
            float s = accI[r];
            for (int j = 0; j <= i; ++j) s += Am[i][j] * Vc[j][d];
            Out[headBase + (size_t)(t0 + i) * (H * 64) + d] = s / den[i];
        }

        // Phase C: state update  S += Kc^T Vc ; z += sum_j k_j.
#pragma unroll
        for (int r = 0; r < 16; ++r) {
            int m = gq * 16 + r;
            float s = Sst[m][d];
            for (int j = 0; j < 64; ++j) s += Kc[j][m] * Vc[j][d];
            Sst[m][d] = s;
        }
        if (tid < 64) {
            float s = zs[tid];
            for (int j = 0; j < 64; ++j) s += Kc[j][tid];
            zs[tid] = s;
        }
        __syncthreads();
    }
}

// ---------------------------------------------------------------------------
// Fused residual + LayerNorm.  One 256-thread block per row of D=1024.
// ---------------------------------------------------------------------------
__global__ __launch_bounds__(256)
void layernorm_res(const float* __restrict__ X, const float* __restrict__ Y,
                   const float* __restrict__ gamma, const float* __restrict__ beta,
                   float* __restrict__ Out)
{
    constexpr int D = 1024;
    __shared__ float red[256], red2[256];
    const int row = blockIdx.x, tid = threadIdx.x;
    const float* xr = X + (size_t)row * D;
    const float* yr = Y + (size_t)row * D;

    float v[4], s = 0.f, s2 = 0.f;
#pragma unroll
    for (int i = 0; i < 4; ++i) {
        int c = tid + i * 256;
        v[i] = xr[c] + yr[c];
        s  += v[i];
        s2 += v[i] * v[i];
    }
    red[tid] = s; red2[tid] = s2;
    __syncthreads();
    for (int off = 128; off > 0; off >>= 1) {
        if (tid < off) { red[tid] += red[tid + off]; red2[tid] += red2[tid + off]; }
        __syncthreads();
    }
    float mu  = red[0] * (1.f / D);
    float var = red2[0] * (1.f / D) - mu * mu;
    float inv = rsqrtf(var + LN_EPS);
#pragma unroll
    for (int i = 0; i < 4; ++i) {
        int c = tid + i * 256;
        Out[(size_t)row * D + c] = (v[i] - mu) * inv * gamma[c] + beta[c];
    }
}

// ---------------------------------------------------------------------------
// Host-side orchestration (graph-capture safe: only kernel launches).
// ---------------------------------------------------------------------------
extern "C" void kernel_launch(void* const* d_in, const int* in_sizes, int n_in,
                              void* d_out, int out_size, void* d_ws, size_t ws_size,
                              hipStream_t stream)
{
    (void)in_sizes; (void)n_in; (void)out_size; (void)ws_size;

    const int L = 3, B = 2, S = 1024, D = 1024, DFF = 4096, H = 16;
    const int Mrows = B * S;   // 2048

    const float* x   = (const float*)d_in[0];
    const float* Wq  = (const float*)d_in[1];
    const float* bq  = (const float*)d_in[2];
    const float* Wk  = (const float*)d_in[3];
    const float* bk  = (const float*)d_in[4];
    const float* Wv  = (const float*)d_in[5];
    const float* bv  = (const float*)d_in[6];
    const float* Wo  = (const float*)d_in[7];
    const float* bo  = (const float*)d_in[8];
    const float* omg = (const float*)d_in[9];
    const float* g1  = (const float*)d_in[10];
    const float* b1  = (const float*)d_in[11];
    const float* g2  = (const float*)d_in[12];
    const float* b2  = (const float*)d_in[13];
    const float* W1  = (const float*)d_in[14];
    const float* bf1 = (const float*)d_in[15];
    const float* W2  = (const float*)d_in[16];
    const float* bf2 = (const float*)d_in[17];
    float* out = (float*)d_out;

    float* ws = (float*)d_ws;
    const size_t tok = (size_t)Mrows * D;      // 2M floats
    float* cur = ws;                           // running activations
    float* qb  = ws + 1 * tok;                 // q / Qf / reused temp
    float* kb  = ws + 2 * tok;                 // k / Kf
    float* vb  = ws + 3 * tok;                 // v
    float* ab  = ws + 4 * tok;                 // attention output
    float* hb  = ws + 5 * tok;                 // post-LN1 hidden
    float* ffn = ws + 6 * tok;                 // [Mrows, DFF] = 8M floats

    const dim3 blk(256);
    const dim3 gD(D / 128, Mrows / 128);       // (8, 16)
    const dim3 gF(DFF / 128, Mrows / 128);     // (32, 16)

    for (int l = 0; l < L; ++l) {
        const float* inp = (l == 0) ? x : cur;

        gemm_wmma<<<gD, blk, 0, stream>>>(inp, Wq + (size_t)l * D * D, bq + (size_t)l * D,
                                          qb, Mrows, D, D, 0);
        gemm_wmma<<<gD, blk, 0, stream>>>(inp, Wk + (size_t)l * D * D, bk + (size_t)l * D,
                                          kb, Mrows, D, D, 0);
        gemm_wmma<<<gD, blk, 0, stream>>>(inp, Wv + (size_t)l * D * D, bv + (size_t)l * D,
                                          vb, Mrows, D, D, 0);

        favor_kernel<<<B * S * H, 64, 0, stream>>>(qb, omg + (size_t)l * 64 * 32);
        favor_kernel<<<B * S * H, 64, 0, stream>>>(kb, omg + (size_t)l * 64 * 32);

        causal_linear_attn<<<B * H, blk, 0, stream>>>(qb, kb, vb, ab);

        gemm_wmma<<<gD, blk, 0, stream>>>(ab, Wo + (size_t)l * D * D, bo + (size_t)l * D,
                                          qb, Mrows, D, D, 0);
        layernorm_res<<<Mrows, blk, 0, stream>>>(inp, qb, g1 + (size_t)l * D, b1 + (size_t)l * D, hb);

        gemm_wmma<<<gF, blk, 0, stream>>>(hb, W1 + (size_t)l * D * DFF, bf1 + (size_t)l * DFF,
                                          ffn, Mrows, DFF, D, 1);
        gemm_wmma<<<gD, blk, 0, stream>>>(ffn, W2 + (size_t)l * DFF * D, bf2 + (size_t)l * D,
                                          qb, Mrows, D, DFF, 0);

        float* o2 = (l == L - 1) ? out : cur;
        layernorm_res<<<Mrows, blk, 0, stream>>>(hb, qb, g2 + (size_t)l * D, b2 + (size_t)l * D, o2);
    }
}
